// MultiHeadAttention_64742337020267
// MI455X (gfx1250) — matrix operand order repaired and checked by reading, not compiled, not run on hardware
//
#include <hip/hip_runtime.h>
#include <hip/hip_bf16.h>

// ---------------------------------------------------------------------------
// MHA forward for MI455X (gfx1250, wave32): bf16 WMMA 16x16x32 everywhere,
// TDM (tensor_load_to_lds) staging for the natural-layout tiles.
// ---------------------------------------------------------------------------

typedef __bf16 bf16_t;
typedef __attribute__((ext_vector_type(16))) __bf16 v16bf;
typedef __attribute__((ext_vector_type(8)))  float  v8f;
typedef __attribute__((ext_vector_type(4))) unsigned uint32x4;
typedef __attribute__((ext_vector_type(8))) int      int32x8;
typedef __attribute__((ext_vector_type(4))) int      int32x4;

#define SEQ   2048
#define NHEAD 16
#define HDIM  64
#define HID   1024
#define QKVLD 3072   // 3*HID
#define ROWS  4096   // B*SEQ

union AFrag { v16bf v; unsigned u[8]; };
union BFrag { v16bf v; uint4 q[2]; };

static __device__ inline v8f vzero8() {
    v8f z = {0.f,0.f,0.f,0.f,0.f,0.f,0.f,0.f};
    return z;
}

static __device__ inline v8f wmma_bf16(v16bf a, v16bf b, v8f c) {
    return __builtin_amdgcn_wmma_f32_16x16x32_bf16(false, a, false, b, (short)0, c, false, false);
}

static __device__ inline float redmax16(float x) {
    #pragma unroll
    for (int m = 1; m < 16; m <<= 1) x = fmaxf(x, __shfl_xor(x, m, 32));
    return x;
}
static __device__ inline float redsum16(float x) {
    #pragma unroll
    for (int m = 1; m < 16; m <<= 1) x += __shfl_xor(x, m, 32);
    return x;
}

// ---------------------------------------------------------------------------
// Tensor Data Mover: 2D bf16 tile load Global->LDS, D# per ISA ch.8.
//   tile_d0 elements per row (contiguous), tile_d1 rows,
//   stride_elems = global row stride, LDS rows padded per pad codes:
//   pad after 2^(pad_int_code+1) DWORDs by (pad_amt_code+1) DWORDs.
// ---------------------------------------------------------------------------
static __device__ inline void tdm_load_tile_bf16(
        unsigned lds_byte_off, const bf16_t* gptr,
        unsigned tile_d0, unsigned tile_d1,
        unsigned tensor_d0, unsigned tensor_d1,
        unsigned long long stride_elems,
        unsigned pad_int_code, unsigned pad_amt_code) {
    unsigned long long ga = (unsigned long long)(size_t)gptr;
    uint32x4 g0;
    g0.x = 1u;                                            // count=1, load, non-gather
    g0.y = lds_byte_off;                                  // lds_addr
    g0.z = (unsigned)(ga & 0xffffffffu);                  // global_addr[31:0]
    g0.w = (unsigned)((ga >> 32) & 0x01ffffffu) | (2u << 30);  // addr[56:32] | type=2

    int32x8 g1;
    g1[0] = (int)((1u << 16)                      // data_size = 2 bytes
                | (1u << 20)                      // pad_enable
                | (pad_int_code << 22)
                | (pad_amt_code << 25));
    g1[1] = (int)((tensor_d0 & 0xffffu) << 16);                                  // dim0[15:0]
    g1[2] = (int)(((tensor_d0 >> 16) & 0xffffu) | ((tensor_d1 & 0xffffu) << 16));// dim0 hi | dim1 lo
    g1[3] = (int)(((tensor_d1 >> 16) & 0xffffu) | ((tile_d0 & 0xffffu) << 16));  // dim1 hi | tile0
    g1[4] = (int)(tile_d1 & 0xffffu);                                            // tile1 | tile2=0
    g1[5] = (int)(unsigned)(stride_elems & 0xffffffffu);                         // dim0_stride lo
    g1[6] = (int)(unsigned)((stride_elems >> 32) & 0xffffu);                     // dim0_stride hi
    g1[7] = 0;

    int32x4 gz = {0, 0, 0, 0};
#if __clang_major__ >= 23
    int32x8 gz8 = {0, 0, 0, 0, 0, 0, 0, 0};
    __builtin_amdgcn_tensor_load_to_lds(g0, g1, gz, gz, gz8, 0);
#else
    __builtin_amdgcn_tensor_load_to_lds(g0, g1, gz, gz, 0);
#endif
}

// ---------------------------------------------------------------------------
// fp32 -> bf16 elementwise convert
// ---------------------------------------------------------------------------
__global__ void cvt_bf16_kernel(const float* __restrict__ in, bf16_t* __restrict__ out, int n) {
    int i = blockIdx.x * blockDim.x + threadIdx.x;
    if (i < n) out[i] = (bf16_t)in[i];
}

// ---------------------------------------------------------------------------
// Tiled bf16 GEMM: C[M][N] = A[M][K] * B[K][N], fp32 accumulate.
// Block: 256 threads (8 waves). Block tile 128(M) x 128(N). K step 32.
// A tile staged by TDM (padded rows); B tile staged transposed by ALU.
// Wave w owns rows 16w..16w+15, all 128 cols (8 accumulators).
// ---------------------------------------------------------------------------
template <bool STORE_BF16>
__global__ __launch_bounds__(256) void gemm_bf16_kernel(
        const bf16_t* __restrict__ A, const bf16_t* __restrict__ B,
        void* __restrict__ Cout, int M, int N, int K) {
    const int n0 = blockIdx.x * 128;
    const int m0 = blockIdx.y * 128;
    const int tid  = threadIdx.x;
    const int wave = tid >> 5;
    const int lane = tid & 31;
    const int ln   = lane & 15;
    const int hf   = lane >> 4;

    // Row pads (+8 bf16 = +16B) kill bank conflicts; A pad matches TDM D# pad.
    __shared__ __align__(16) bf16_t Asub[128][40];   // [m][k], 80B rows
    __shared__ __align__(16) bf16_t Bt[128][40];     // [n][k], transposed

    v8f acc[8];
    #pragma unroll
    for (int t = 0; t < 8; ++t) acc[t] = vzero8();

    for (int k0 = 0; k0 < K; k0 += 32) {
        __syncthreads();
        // --- A tile 128x32 via Tensor Data Mover (rows 64B, pad 16B) ---
        if (wave == 0) {
            tdm_load_tile_bf16((unsigned)(size_t)(void*)&Asub[0][0],
                               A + (size_t)m0 * K + k0,
                               /*tile*/ 32, 128, /*tensor*/ 32, 128,
                               (unsigned long long)K,
                               /*pad: every 16 DW*/ 3, /*by 4 DW*/ 3);
        }
        // --- B tile 32x128 transposed (each thread: 16 contiguous cols) ---
        {
            int flat = tid * 16;
            int kk = flat >> 7;
            int nn = flat & 127;
            const uint4* src = reinterpret_cast<const uint4*>(B + (size_t)(k0 + kk) * N + n0 + nn);
            union { uint4 q[2]; bf16_t e[16]; } u;
            u.q[0] = src[0]; u.q[1] = src[1];
            #pragma unroll
            for (int i = 0; i < 16; ++i) Bt[nn + i][kk] = u.e[i];
        }
        if (wave == 0) __builtin_amdgcn_s_wait_tensorcnt(0);
        __syncthreads();

        // --- A fragment: row = lane%16, contiguous K (16*half + s) ---
        AFrag af;
        #pragma unroll
        for (int v = 0; v < 8; ++v) {
            int kp = 2 * v + 16 * hf;
            af.u[v] = *reinterpret_cast<const unsigned*>(&Asub[wave * 16 + ln][kp]);
        }
        // --- preload all 8 B fragments, then back-to-back WMMAs ---
        BFrag bfg[8];
        #pragma unroll
        for (int nt = 0; nt < 8; ++nt) {
            const uint4* p = reinterpret_cast<const uint4*>(&Bt[16 * nt + ln][16 * hf]);
            bfg[nt].q[0] = p[0]; bfg[nt].q[1] = p[1];
        }
        #pragma unroll
        for (int nt = 0; nt < 8; ++nt)
            acc[nt] = wmma_bf16(af.v, bfg[nt].v, acc[nt]);
    }

    // --- epilogue: C layout row = v + 8*half, col = lane%16 + 16*nt ---
    #pragma unroll
    for (int v = 0; v < 8; ++v) {
        int r = m0 + wave * 16 + v + 8 * hf;
        #pragma unroll
        for (int nt = 0; nt < 8; ++nt) {
            int c = n0 + 16 * nt + ln;
            if (STORE_BF16) ((bf16_t*)Cout)[(size_t)r * N + c] = (bf16_t)acc[nt][v];
            else            ((float*)Cout)[(size_t)r * N + c]  = acc[nt][v];
        }
    }
}

// ---------------------------------------------------------------------------
// In-place RoPE on q and k halves of the bf16 qkv buffer.
// qkv layout: [B*S][ h*192 + {q:0..63 | k:64..127 | v:128..191} ]
// ---------------------------------------------------------------------------
__global__ void rope_kernel(bf16_t* __restrict__ qkv) {
    int idx = blockIdx.x * blockDim.x + threadIdx.x;  // 2*2048*16*32 = 2^21
    int j = idx & 31;
    int h = (idx >> 5) & 15;
    int s = (idx >> 9) & 2047;
    int b = idx >> 20;
    size_t base = ((size_t)(b * SEQ + s)) * QKVLD + h * 192;

    float inv_freq = __powf(10000.f, -(float)(2 * j) * (1.f / 64.f));
    float ang = (float)s * inv_freq;
    float c  = __cosf(ang);
    float sn = __sinf(ang);

    float q0 = (float)qkv[base + j];
    float q1 = (float)qkv[base + j + 32];
    qkv[base + j]      = (bf16_t)(q0 * c - q1 * sn);
    qkv[base + j + 32] = (bf16_t)(q1 * c + q0 * sn);

    float k0 = (float)qkv[base + 64 + j];
    float k1 = (float)qkv[base + 96 + j];
    qkv[base + 64 + j] = (bf16_t)(k0 * c - k1 * sn);
    qkv[base + 96 + j] = (bf16_t)(k1 * c + k0 * sn);
}

// ---------------------------------------------------------------------------
// Flash-style attention. Block = (b, h, 128-query tile); 8 waves, each owns a
// 16-query strip and the full 64-dim output. Keys processed 32 at a time;
// K tile staged by TDM (natural [key][d] layout), V tile transposed by ALU.
// ---------------------------------------------------------------------------
__global__ __launch_bounds__(256) void attn_kernel(
        const bf16_t* __restrict__ qkv, const int* __restrict__ mask,
        bf16_t* __restrict__ ctx) {
    const int blk   = blockIdx.x;          // 512 = B * NHEAD * (SEQ/128)
    const int qtile = blk & 15;
    const int h     = (blk >> 4) & 15;
    const int b     = blk >> 8;
    const int tid   = threadIdx.x;
    const int wave  = tid >> 5;
    const int lane  = tid & 31;
    const int ln    = lane & 15;
    const int hf    = lane >> 4;

    __shared__ __align__(16) bf16_t Klds[32][72];     // [key][d], 144B rows (TDM pad)
    __shared__ __align__(16) bf16_t Vt[64][40];       // [d][key], 80B rows
    __shared__ __align__(16) bf16_t Plds[8][16][36];  // per-wave P bounce, 72B rows

    const int    qbase    = qtile * 128;
    const size_t base_row = (size_t)b * SEQ;

    // --- Q fragments (2 x 16x32), loaded once straight from global ---
    AFrag qf[2];
    {
        int qrow = qbase + wave * 16 + ln;             // A layout: row = lane%16
        const bf16_t* qptr = qkv + (base_row + qrow) * (size_t)QKVLD + h * 192;
        #pragma unroll
        for (int f = 0; f < 2; ++f) {
            #pragma unroll
            for (int v = 0; v < 8; ++v) {
                int d = 32 * f + 2 * v + 16 * hf;
                qf[f].u[v] = *reinterpret_cast<const unsigned*>(qptr + d);
            }
        }
    }

    v8f acc[4];
    #pragma unroll
    for (int t = 0; t < 4; ++t) acc[t] = vzero8();
    float rmax[8], rsum[8];
    #pragma unroll
    for (int v = 0; v < 8; ++v) { rmax[v] = -3.0e38f; rsum[v] = 0.f; }

    for (int kb = 0; kb < SEQ; kb += 32) {
        __syncthreads();
        // --- K tile 32x64 via TDM (row 128B, pad 16B -> 144B LDS rows) ---
        if (wave == 0) {
            tdm_load_tile_bf16((unsigned)(size_t)(void*)&Klds[0][0],
                               qkv + (base_row + kb) * (size_t)QKVLD + h * 192 + 64,
                               /*tile*/ 64, 32, /*tensor*/ 64, 32,
                               (unsigned long long)QKVLD,
                               /*pad: every 32 DW*/ 4, /*by 4 DW*/ 3);
        }
        // --- V tile staged transposed by all threads ---
        {
            int flat = tid * 8;
            int key  = flat >> 6;
            int d    = flat & 63;
            const bf16_t* vp = qkv + (base_row + kb + key) * (size_t)QKVLD + h * 192 + 128 + d;
            union { uint4 q; bf16_t e[8]; } vu;
            vu.q = *reinterpret_cast<const uint4*>(vp);
            #pragma unroll
            for (int i = 0; i < 8; ++i) Vt[d + i][key] = vu.e[i];
        }
        if (wave == 0) __builtin_amdgcn_s_wait_tensorcnt(0);
        __syncthreads();

        // mask bias per key column of each 16-wide score tile
        float bias0 = (mask[base_row + kb + ln]      == 0) ? -1.0e30f : 0.f;
        float bias1 = (mask[base_row + kb + 16 + ln] == 0) ? -1.0e30f : 0.f;

        // --- scores S = Q K^T (two 16x16 tiles, two K-steps each) ---
        v8f s0 = vzero8(), s1 = vzero8();
        #pragma unroll
        for (int ks = 0; ks < 2; ++ks) {
            BFrag kf0, kf1;
            const uint4* p0 = reinterpret_cast<const uint4*>(&Klds[ln][32 * ks + 16 * hf]);
            const uint4* p1 = reinterpret_cast<const uint4*>(&Klds[16 + ln][32 * ks + 16 * hf]);
            kf0.q[0] = p0[0]; kf0.q[1] = p0[1];
            kf1.q[0] = p1[0]; kf1.q[1] = p1[1];
            s0 = wmma_bf16(qf[ks].v, kf0.v, s0);
            s1 = wmma_bf16(qf[ks].v, kf1.v, s1);
        }

        // --- online softmax (row = v + 8*half spread over 16 lanes) ---
        const float sc = 0.125f;   // 1/sqrt(64)
        float nmax[8];
        #pragma unroll
        for (int v = 0; v < 8; ++v) {
            float a = s0[v] * sc + bias0;
            float c = s1[v] * sc + bias1;
            s0[v] = a; s1[v] = c;
            float m = redmax16(fmaxf(a, c));
            nmax[v] = fmaxf(rmax[v], m);
        }
        #pragma unroll
        for (int v = 0; v < 8; ++v) {
            float scale = __expf(rmax[v] - nmax[v]);
            rmax[v] = nmax[v];
            float p0 = __expf(s0[v] - nmax[v]);
            float p1 = __expf(s1[v] - nmax[v]);
            rsum[v] = rsum[v] * scale + redsum16(p0 + p1);
            Plds[wave][v + 8 * hf][ln]      = (bf16_t)p0;
            Plds[wave][v + 8 * hf][16 + ln] = (bf16_t)p1;
            acc[0][v] *= scale; acc[1][v] *= scale;
            acc[2][v] *= scale; acc[3][v] *= scale;
        }

        // --- P back out of LDS in A-fragment layout (wave-local, DS in-order) ---
        AFrag pf;
        #pragma unroll
        for (int v = 0; v < 8; ++v) {
            int k = 2 * v + 16 * hf;
            pf.u[v] = *reinterpret_cast<const unsigned*>(&Plds[wave][ln][k]);
        }

        // --- O += P V (4 output tiles across head dim) ---
        #pragma unroll
        for (int nt = 0; nt < 4; ++nt) {
            BFrag vf;
            const uint4* p = reinterpret_cast<const uint4*>(&Vt[16 * nt + ln][16 * hf]);
            vf.q[0] = p[0]; vf.q[1] = p[1];
            acc[nt] = wmma_bf16(pf.v, vf.v, acc[nt]);
        }
    }

    // --- normalize and write context [B*S][HID] (head-interleaved) ---
    #pragma unroll
    for (int v = 0; v < 8; ++v) {
        float inv = 1.f / rsum[v];
        int qrow = qbase + wave * 16 + v + 8 * hf;
        size_t rowoff = (base_row + qrow) * (size_t)HID + h * HDIM;
        #pragma unroll
        for (int nt = 0; nt < 4; ++nt)
            ctx[rowoff + 16 * nt + ln] = (bf16_t)(acc[nt][v] * inv);
    }
}

// ---------------------------------------------------------------------------
// Host-side launch
// ---------------------------------------------------------------------------
extern "C" void kernel_launch(void* const* d_in, const int* in_sizes, int n_in,
                              void* d_out, int out_size, void* d_ws, size_t ws_size,
                              hipStream_t stream) {
    const float* x     = (const float*)d_in[0];   // [2,2048,1024]
    const int*   amask = (const int*)d_in[1];     // [2,2048]
    const float* wqkv  = (const float*)d_in[2];   // [1024,3072]
    const float* wout  = (const float*)d_in[3];   // [1024,1024]
    float*       out   = (float*)d_out;           // [2,2048,1024]

    char* ws = (char*)d_ws;
    bf16_t* x_bf    = (bf16_t*)ws; ws += (size_t)ROWS * HID * 2;
    bf16_t* wqkv_bf = (bf16_t*)ws; ws += (size_t)HID * QKVLD * 2;
    bf16_t* wout_bf = (bf16_t*)ws; ws += (size_t)HID * HID * 2;
    bf16_t* qkv_bf  = (bf16_t*)ws; ws += (size_t)ROWS * QKVLD * 2;
    bf16_t* ctx_bf  = (bf16_t*)ws; ws += (size_t)ROWS * HID * 2;

    // fp32 -> bf16 converts
    {
        int n;
        n = ROWS * HID;
        cvt_bf16_kernel<<<(n + 255) / 256, 256, 0, stream>>>(x, x_bf, n);
        n = HID * QKVLD;
        cvt_bf16_kernel<<<(n + 255) / 256, 256, 0, stream>>>(wqkv, wqkv_bf, n);
        n = HID * HID;
        cvt_bf16_kernel<<<(n + 255) / 256, 256, 0, stream>>>(wout, wout_bf, n);
    }

    // qkv = x @ w_qkv  (bf16 out)
    gemm_bf16_kernel<true><<<dim3(QKVLD / 128, ROWS / 128), 256, 0, stream>>>(
        x_bf, wqkv_bf, (void*)qkv_bf, ROWS, QKVLD, HID);

    // RoPE in place on q,k
    rope_kernel<<<(2 * SEQ * NHEAD * 32) / 256, 256, 0, stream>>>(qkv_bf);

    // attention -> context (bf16)
    attn_kernel<<<2 * NHEAD * (SEQ / 128), 256, 0, stream>>>(qkv_bf, amask, ctx_bf);

    // out = context @ w_out  (fp32 out)
    gemm_bf16_kernel<false><<<dim3(HID / 128, ROWS / 128), 256, 0, stream>>>(
        ctx_bf, wout_bf, (void*)out, ROWS, HID, HID);
}
